// GPTAttention_75359496176084
// MI455X (gfx1250) — matrix operand order repaired and checked by reading, not compiled, not run on hardware
//
#include <hip/hip_runtime.h>
#include <float.h>

typedef __attribute__((ext_vector_type(16))) __bf16 bf16x16;
typedef __attribute__((ext_vector_type(8)))  __bf16 bf16x8;
typedef __attribute__((ext_vector_type(8)))  float   f32x8;
typedef unsigned short u16;
typedef __attribute__((ext_vector_type(4))) u16 u16x4;
typedef __attribute__((ext_vector_type(8))) u16 u16x8;

#define ASTR 40   // 32 bf16 + 8 pad; rows stay 16B aligned (80B)

static __device__ __forceinline__ u16 f2bf(float x) {
    unsigned u = __builtin_bit_cast(unsigned, x);
    u += 0x7FFFu + ((u >> 16) & 1u);            // round-to-nearest-even
    return (u16)(u >> 16);
}

static __device__ __forceinline__ bf16x16 frag16(const u16* p0, const u16* p1) {
    bf16x8 lo = *(const bf16x8*)p0;
    bf16x8 hi = *(const bf16x8*)p1;
    return __builtin_shufflevector(lo, hi, 0,1,2,3,4,5,6,7,8,9,10,11,12,13,14,15);
}

template<bool P, typename T, typename F> struct cond      { using type = T; };
template<typename T, typename F>         struct cond<false, T, F> { using type = F; };

// ---- staging: row-major [ROWS x 32] source -> lds[row][k] (bf16) ----
template<typename T, int ROWS> struct StageRM;

template<int ROWS> struct StageRM<float, ROWS> {
    static constexpr int N = ROWS * 8 / 256;
    float4 r[N];
    __device__ __forceinline__ void load(const float* g, long long ld, int tid) {
        #pragma unroll
        for (int i = 0; i < N; ++i) {
            int q = tid + i * 256;
            r[i] = *(const float4*)(g + (long long)(q >> 3) * ld + (q & 7) * 4);
        }
    }
    __device__ __forceinline__ void store(u16* lds, int tid) {
        #pragma unroll
        for (int i = 0; i < N; ++i) {
            int q = tid + i * 256;
            u16x4 w = { f2bf(r[i].x), f2bf(r[i].y), f2bf(r[i].z), f2bf(r[i].w) };
            *(u16x4*)&lds[(q >> 3) * ASTR + (q & 7) * 4] = w;
        }
    }
};

template<int ROWS> struct StageRM<u16, ROWS> {       // bf16 source: straight copy
    static constexpr int N = ROWS * 4 / 256;
    u16x8 r[N];
    __device__ __forceinline__ void load(const u16* g, long long ld, int tid) {
        #pragma unroll
        for (int i = 0; i < N; ++i) {
            int q = tid + i * 256;
            r[i] = *(const u16x8*)(g + (long long)(q >> 2) * ld + (q & 3) * 8);
        }
    }
    __device__ __forceinline__ void store(u16* lds, int tid) {
        #pragma unroll
        for (int i = 0; i < N; ++i) {
            int q = tid + i * 256;
            *(u16x8*)&lds[(q >> 2) * ASTR + (q & 3) * 8] = r[i];
        }
    }
};

// ---- staging: row-major [32 x BN] source -> transposed lds[n][k] (bf16) ----
template<typename T, int BN> struct StageTR;

template<int BN> struct StageTR<float, BN> {
    static constexpr int N = BN * 8 / 256;
    float4 r[N];
    __device__ __forceinline__ void load(const float* g, long long ld, int tid) {
        #pragma unroll
        for (int i = 0; i < N; ++i) {
            int q = tid + i * 256;
            r[i] = *(const float4*)(g + (long long)(q / (BN / 4)) * ld + (q % (BN / 4)) * 4);
        }
    }
    __device__ __forceinline__ void store(u16* lds, int tid) {
        #pragma unroll
        for (int i = 0; i < N; ++i) {
            int q = tid + i * 256;
            int kk = q / (BN / 4), n4 = (q % (BN / 4)) * 4;
            lds[(n4 + 0) * ASTR + kk] = f2bf(r[i].x);
            lds[(n4 + 1) * ASTR + kk] = f2bf(r[i].y);
            lds[(n4 + 2) * ASTR + kk] = f2bf(r[i].z);
            lds[(n4 + 3) * ASTR + kk] = f2bf(r[i].w);
        }
    }
};

template<int BN> struct StageTR<u16, BN> {
    static constexpr int N = BN / 64;
    u16x8 r[N];
    __device__ __forceinline__ void load(const u16* g, long long ld, int tid) {
        #pragma unroll
        for (int i = 0; i < N; ++i) {
            int q = tid + i * 256;
            r[i] = *(const u16x8*)(g + (long long)(q / (BN / 8)) * ld + (q % (BN / 8)) * 8);
        }
    }
    __device__ __forceinline__ void store(u16* lds, int tid) {
        #pragma unroll
        for (int i = 0; i < N; ++i) {
            int q = tid + i * 256;
            int kk = q / (BN / 8), n8 = (q % (BN / 8)) * 8;
            #pragma unroll
            for (int j = 0; j < 8; ++j) lds[(n8 + j) * ASTR + kk] = r[i][j];
        }
    }
};

// ---- C-side helpers ----
static __device__ __forceinline__ void cstore(float* p, float v) { *p = v; }
static __device__ __forceinline__ void cstore(u16* p, float v)   { *p = f2bf(v); }
static __device__ __forceinline__ void mask4(float* p) {
    *(float4*)p = make_float4(-FLT_MAX, -FLT_MAX, -FLT_MAX, -FLT_MAX);
}
static __device__ __forceinline__ void mask4(u16* p) {
    u16 m = f2bf(-FLT_MAX); u16x4 w = { m, m, m, m }; *(u16x4*)p = w;
}

// bf16-WMMA GEMM, double-buffered LDS, 1 barrier per 32-deep K-step.
// Block tile 128 x (WNT*32); 8 waves as 4(M) x 2(N); wave tile 32 x (WNT*16).
// TA/TB/TC in {float, u16(bf16)}. If B_IS_T, B is B^T row-major [N][K] (ldb),
// else row-major [K][N] (transposed during staging). Batched over blockIdx.z:
// z -> bz=z/Hd, hz=z%Hd with separate element strides.
template<typename TA, typename TB, typename TC, bool B_IS_T, int WNT>
__global__ __launch_bounds__(256) void wmma_gemm(
    const TA* __restrict__ A, const TB* __restrict__ B,
    TC* __restrict__ C, const float* __restrict__ bias,
    int K, int lda, int ldb, int ldc, int Hd,
    long long sAb, long long sAh, long long sBb, long long sBh,
    long long sCb, long long sCh, float scale, int causal)
{
    constexpr int BN  = WNT * 32;        // 128 or 64
    constexpr int BN4 = BN / 4;

    __shared__ u16 a_lds[2][128 * ASTR];
    __shared__ u16 b_lds[2][BN * ASTR];

    const int z  = blockIdx.z;
    const int bz = z / Hd, hz = z % Hd;
    A += (long long)bz * sAb + (long long)hz * sAh;
    B += (long long)bz * sBb + (long long)hz * sBh;
    C += (long long)bz * sCb + (long long)hz * sCh;

    const int tid = threadIdx.x;
    const int bm  = blockIdx.y * 128;
    const int bn  = blockIdx.x * BN;

    if (causal && bn > bm + 127) {                 // tile entirely above diagonal
        #pragma unroll
        for (int i = 0; i < (128 * BN) / 1024; ++i) {
            int q = tid + i * 256;                 // 128*BN/4 quads
            int r = q / BN4, c4 = (q % BN4) * 4;
            mask4(C + (long long)(bm + r) * ldc + bn + c4);
        }
        return;
    }

    const int lane = tid & 31;
    const int wid  = tid >> 5;                     // 8 waves: 4 (M) x 2 (N)
    const int wm   = (wid & 3) * 32;
    const int wn   = (wid >> 2) * (WNT * 16);
    const int hi   = lane >> 4;
    const int l16  = lane & 15;

    f32x8 acc[2][WNT];
    #pragma unroll
    for (int mi = 0; mi < 2; ++mi)
        #pragma unroll
        for (int f = 0; f < WNT; ++f) acc[mi][f] = {};

    using ASt = StageRM<TA, 128>;
    using BSt = typename cond<B_IS_T, StageRM<TB, BN>, StageTR<TB, BN> >::type;
    ASt sa;
    BSt sb;

    const TA* Abase = A + (long long)bm * lda;
    const int nsteps = K >> 5;

    // prologue: tile 0
    sa.load(Abase, lda, tid);
    sb.load(B_IS_T ? (B + (long long)bn * ldb) : (B + bn), ldb, tid);
    sa.store(a_lds[0], tid);
    sb.store(b_lds[0], tid);
    __syncthreads();

    for (int i = 0; i < nsteps; ++i) {
        const int cur  = i & 1;
        const bool more = (i + 1) < nsteps;

        if (more) {                                // prefetch tile i+1 (overlaps WMMAs)
            const int k0 = (i + 1) << 5;
            sa.load(Abase + k0, lda, tid);
            sb.load(B_IS_T ? (B + (long long)bn * ldb + k0)
                           : (B + (long long)k0 * ldb + bn), ldb, tid);
        }

        // compute from lds[cur]
        {
            const u16* al = a_lds[cur];
            const u16* bl = b_lds[cur];
            bf16x16 af[2];
            #pragma unroll
            for (int mi = 0; mi < 2; ++mi) {
                int am = wm + mi * 16 + l16;       // A: m = lane&15 (both halves)
                af[mi] = frag16(&al[am * ASTR + hi * 8], &al[am * ASTR + 16 + hi * 8]);
            }
            bf16x16 bfv[WNT];
            #pragma unroll
            for (int f = 0; f < WNT; ++f) {
                int n = wn + f * 16 + l16;         // B: k = hi*8 + e (e<8), 16 + hi*8 + (e-8) (e>=8)
                bfv[f] = frag16(&bl[n * ASTR + hi * 8], &bl[n * ASTR + 16 + hi * 8]);
            }
            #pragma unroll
            for (int mi = 0; mi < 2; ++mi)
                #pragma unroll
                for (int f = 0; f < WNT; ++f)
                    acc[mi][f] = __builtin_amdgcn_wmma_f32_16x16x32_bf16(
                        false, af[mi], false, bfv[f], (short)0, acc[mi][f], false, false);
        }

        if (more) {                                // stage tile i+1 into lds[cur^1]
            sa.store(a_lds[cur ^ 1], tid);
            sb.store(b_lds[cur ^ 1], tid);
        }
        __syncthreads();
    }

    // store (D layout: row = e + hi*8, col = lane&15)
    #pragma unroll
    for (int mi = 0; mi < 2; ++mi) {
        const int gr0 = bm + wm + mi * 16 + hi * 8;
        #pragma unroll
        for (int f = 0; f < WNT; ++f) {
            const int gc  = bn + wn + f * 16 + l16;
            const float bia = bias ? bias[gc] : 0.0f;
            #pragma unroll
            for (int e = 0; e < 8; ++e) {
                int gr = gr0 + e;
                float v = acc[mi][f][e] * scale + bia;
                if (causal && gc > gr) v = -FLT_MAX;
                cstore(C + (long long)gr * ldc + gc, v);
            }
        }
    }
}

// In-place softmax over rows of 2048 (one 256-thread block per row)
__global__ __launch_bounds__(256) void softmax_rows(float* __restrict__ P)
{
    const long long row = blockIdx.x;
    float* p = P + row * 2048LL;
    __shared__ float red[8];
    const int t = threadIdx.x, lane = t & 31, wid = t >> 5;

    float v[8];
    float4 x0 = *(const float4*)(p + t * 4);
    float4 x1 = *(const float4*)(p + 1024 + t * 4);
    v[0]=x0.x; v[1]=x0.y; v[2]=x0.z; v[3]=x0.w;
    v[4]=x1.x; v[5]=x1.y; v[6]=x1.z; v[7]=x1.w;

    float m = v[0];
    #pragma unroll
    for (int e = 1; e < 8; ++e) m = fmaxf(m, v[e]);
    #pragma unroll
    for (int off = 16; off > 0; off >>= 1) m = fmaxf(m, __shfl_xor(m, off));
    if (lane == 0) red[wid] = m;
    __syncthreads();
    float bm = red[0];
    #pragma unroll
    for (int i = 1; i < 8; ++i) bm = fmaxf(bm, red[i]);
    __syncthreads();

    float s = 0.0f;
    #pragma unroll
    for (int e = 0; e < 8; ++e) { v[e] = __expf(v[e] - bm); s += v[e]; }
    #pragma unroll
    for (int off = 16; off > 0; off >>= 1) s += __shfl_xor(s, off);
    if (lane == 0) red[wid] = s;
    __syncthreads();
    float bs = 0.0f;
    #pragma unroll
    for (int i = 0; i < 8; ++i) bs += red[i];
    float inv = 1.0f / bs;

    float4 y0 = make_float4(v[0]*inv, v[1]*inv, v[2]*inv, v[3]*inv);
    float4 y1 = make_float4(v[4]*inv, v[5]*inv, v[6]*inv, v[7]*inv);
    *(float4*)(p + t * 4) = y0;
    *(float4*)(p + 1024 + t * 4) = y1;
}

extern "C" void kernel_launch(void* const* d_in, const int* in_sizes, int n_in,
                              void* d_out, int out_size, void* d_ws, size_t ws_size,
                              hipStream_t stream) {
    const float* X  = (const float*)d_in[0];   // [4096,1024]
    const float* Wf = (const float*)d_in[1];   // [1024,3072]
    const float* Wp = (const float*)d_in[2];   // [1024,1024]
    const float* bp = (const float*)d_in[3];   // [1024]
    float* out   = (float*)d_out;              // attn_out [4096,1024] then weights [32,2048,2048]
    float* attnw = out + 4096LL * 1024;
    u16*   qkv   = (u16*)d_ws;                 // [4096,3072] bf16
    u16*   ao    = qkv + 4096LL * 3072;        // [4096,1024] bf16

    dim3 blk(256);

    // 1) QKV = X @ W_fused -> bf16   (M=4096, N=3072, K=1024)
    wmma_gemm<float, float, u16, false, 4><<<dim3(3072/128, 4096/128, 1), blk, 0, stream>>>(
        X, Wf, qkv, nullptr,
        1024, 1024, 3072, 3072, 1,
        0, 0, 0, 0, 0, 0, 1.0f, 0);

    // 2) scores = Q K^T / 8 (bf16 operands), causal -> d_out weights (per b,h: 2048^2, K=64)
    wmma_gemm<u16, u16, float, true, 4><<<dim3(2048/128, 2048/128, 32), blk, 0, stream>>>(
        qkv /*Q*/, qkv + 1024 /*K*/, attnw, nullptr,
        64, 3072, 3072, 2048, 16,
        2048LL*3072, 64, 2048LL*3072, 64,
        16LL*2048*2048, 2048LL*2048,
        0.125f, 1);

    // 3) softmax in place (65536 rows of 2048)
    softmax_rows<<<dim3(65536), blk, 0, stream>>>(attnw);

    // 4) O = P(f32) @ V(bf16) -> bf16 ao in (b, n, h*64+d) layout (M=2048, N=64, K=2048)
    wmma_gemm<float, u16, u16, false, 2><<<dim3(1, 2048/128, 32), blk, 0, stream>>>(
        attnw, qkv + 2048 /*V*/, ao, nullptr,
        2048, 2048, 3072, 1024, 16,
        16LL*2048*2048, 2048LL*2048,
        2048LL*3072, 64,
        2048LL*1024, 64,
        1.0f, 0);

    // 5) out = ao(bf16) @ W_proj + b_proj   (M=4096, N=1024, K=1024)
    wmma_gemm<u16, float, float, false, 4><<<dim3(1024/128, 4096/128, 1), blk, 0, stream>>>(
        ao, Wp, out, bp,
        1024, 1024, 1024, 1024, 1,
        0, 0, 0, 0, 0, 0, 1.0f, 0);
}
